// APDNC_60842506715774
// MI455X (gfx1250) — compile-verified, matrix operand order checked by reading, not run on hardware
//
#include <hip/hip_runtime.h>
#include <stdint.h>

// ---------------- problem constants ----------------
constexpr int kBS = 32, kT = 32, kXD = 128, kHD = 512;
constexpr int kG4 = 4 * kHD;          // 2048 gates per direction
constexpr int kVT = 2048, kW = 64, kR = 4, kN = 1024;
constexpr int kET = 463, kETP = 464;  // interface vector (padded to x16)
constexpr int kIN0 = 384;             // xd + W*R
constexpr int kFLAT = 2048;           // 2*L*hdim
constexpr int kKC = 512;              // K-chunk staged in LDS per GEMM block

// interface vector offsets (reference parse order)
constexpr int kOffRkeys = 0;    // [0,256)   rkeys[b,w,r] = iv[w*R+r]
constexpr int kOffRbeta = 256;  // [256,260)
constexpr int kOffWkey  = 260;  // [260,324)
constexpr int kOffWbeta = 324;
constexpr int kOffErase = 325;  // [325,389)
constexpr int kOffWvec  = 389;  // [389,453)
constexpr int kOffFg    = 453;  // [453,457)
constexpr int kOffAg    = 457;
constexpr int kOffWg    = 458;

constexpr float kEPS = 1e-8f;

// ---------------- CDNA5 types ----------------
typedef __attribute__((ext_vector_type(16))) __bf16 v16bf;
typedef __attribute__((ext_vector_type(8)))  float  v8f;
typedef __attribute__((ext_vector_type(4)))  unsigned int v4u;
typedef __attribute__((ext_vector_type(8)))  int v8i;
typedef __attribute__((ext_vector_type(4)))  int v4i;
union Frag { v16bf v; uint32_t u[8]; uint4 q[2]; };

__device__ __forceinline__ uint16_t f2bf(float f) {
  uint32_t x = __builtin_bit_cast(uint32_t, f);
  x += 0x7FFFu + ((x >> 16) & 1u);      // round-to-nearest-even
  return (uint16_t)(x >> 16);
}
__device__ __forceinline__ float sigm(float x) { return 1.0f / (1.0f + __expf(-x)); }
__device__ __forceinline__ float softplusf(float x) {
  return fmaxf(x, 0.0f) + log1pf(__expf(-fabsf(x)));
}

// CDNA5 async global->LDS staging (ASYNCcnt tracked).  Low 32b of a flat LDS
// pointer are the LDS byte offset (ISA aperture rules), which is what VDST wants.
__device__ __forceinline__ void async_ld_f32(float* lds_dst, const float* g) {
  uint32_t l = (uint32_t)(size_t)lds_dst;
  unsigned long long ga = (unsigned long long)(size_t)g;
  asm volatile("global_load_async_to_lds_b32 %0, %1, off" :: "v"(l), "v"(ga) : "memory");
}
__device__ __forceinline__ void async_ld_b128(uint16_t* lds_dst, const uint16_t* g) {
  uint32_t l = (uint32_t)(size_t)lds_dst;
  unsigned long long ga = (unsigned long long)(size_t)g;
  asm volatile("global_load_async_to_lds_b128 %0, %1, off" :: "v"(l), "v"(ga) : "memory");
}
__device__ __forceinline__ void wait_async0() {
  asm volatile("s_wait_asynccnt 0" ::: "memory");
}

// ---------------- bf16 WMMA GEMM ----------------
// C(32 x Nout) = A(32 x K, stride lda, bf16) * B(Nout x K, bf16 row-major)^T [+bias] [+=C]
// A panel is staged cooperatively into LDS in kKC-column chunks via async
// global->LDS copies (shared by all 8 waves of the block); B streams from L2
// as two b128 fragment loads per wave per K-step; 2 WMMAs cover M=32.
__global__ void __launch_bounds__(256)
wmma_gemm_kernel(const uint16_t* __restrict__ A, int lda,
                 const uint16_t* __restrict__ B,
                 const float* __restrict__ bias,
                 float* __restrict__ C, int ldc,
                 int K, int Nout, int nValid, int accum) {
  __shared__ uint16_t sA[32 * (kKC + 8)];          // +8 elems/row: bank-conflict pad
  const int tid  = threadIdx.x;
  const int wave = (blockIdx.x * blockDim.x + tid) >> 5;
  const int lane = tid & 31;
  const int mn   = lane & 15;   // m (A) / n (B,C,D) within tile
  const int hi   = lane >> 4;

  const int n0 = wave * 16;
  const bool valid = (n0 < Nout);
  const int n0c = valid ? n0 : (Nout - 16);        // clamp: keep barriers uniform

  const uint16_t* Bp  = B + (size_t)(n0c + mn) * K;
  const uint16_t* sA0 = sA + (size_t)mn * (kKC + 8);
  const uint16_t* sA1 = sA + (size_t)(mn + 16) * (kKC + 8);

  v8f acc0 = {}; v8f acc1 = {};
  for (int kc = 0; kc < K; kc += kKC) {
    const int kcLen = (K - kc < kKC) ? (K - kc) : kKC;
    __syncthreads();                               // previous chunk fully consumed
    const int upr = kcLen >> 3;                    // 16B units per row
    for (int u = tid; u < 32 * upr; u += 256) {
      int m = u / upr, c2 = u - m * upr;
      async_ld_b128(sA + (size_t)m * (kKC + 8) + (c2 << 3),
                    A + (size_t)m * lda + kc + (c2 << 3));
    }
    wait_async0();                                 // own wave's copies done
    __syncthreads();                               // everyone's copies visible

    for (int k0 = 0; k0 < kcLen; k0 += 32) {
      if (k0 + 32 < kcLen) __builtin_prefetch(Bp + kc + k0 + 32, 0, 1);
      Frag a0, a1, b;
      const int kl = k0 + (hi << 3);               // fragment = 2 x 16B runs per lane
      a0.q[0] = *(const uint4*)(sA0 + kl);
      a0.q[1] = *(const uint4*)(sA0 + kl + 16);
      a1.q[0] = *(const uint4*)(sA1 + kl);
      a1.q[1] = *(const uint4*)(sA1 + kl + 16);
      const int kg = kc + kl;
      b.q[0] = *(const uint4*)(Bp + kg);
      b.q[1] = *(const uint4*)(Bp + kg + 16);
      acc0 = __builtin_amdgcn_wmma_f32_16x16x32_bf16(false, a0.v, false, b.v, (short)0, acc0, false, false);
      acc1 = __builtin_amdgcn_wmma_f32_16x16x32_bf16(false, a1.v, false, b.v, (short)0, acc1, false, false);
    }
  }

  const int ncol = n0 + mn;
  if (valid && ncol < nValid) {
    float bv = bias ? bias[ncol] : 0.0f;
#pragma unroll
    for (int r = 0; r < 8; ++r) {
      int m = r + (hi << 3);                       // C/D: vgpr r -> M=r / M=r+8
      float v0 = acc0[r] + bv;
      float v1 = acc1[r] + bv;
      float* c0 = C + (size_t)m * ldc + ncol;
      float* c1 = C + (size_t)(m + 16) * ldc + ncol;
      if (accum) { *c0 += v0; *c1 += v1; } else { *c0 = v0; *c1 = v1; }
    }
  }
}

// ---------------- TDM state copy (memory0 -> M) ----------------
// One wave per block DMAs a 2048x8-float tile global->LDS->global via the
// Tensor Data Mover.  D# built per ISA ch.8: group0 {count, lds_addr,
// global_addr, type=2}, group1 {data_size=4B, tensor/tile dims, stride}.
__global__ void __launch_bounds__(32) tdm_copy_kernel(const float* __restrict__ src,
                                                      float* __restrict__ dst) {
  __shared__ float tile[2048 * 8];                 // 64KB
  const size_t ofs = (size_t)blockIdx.x * (2048 * 8);
  unsigned long long ga = (unsigned long long)(size_t)(src + ofs);
  unsigned long long gb = (unsigned long long)(size_t)(dst + ofs);
  unsigned int lds = (unsigned int)(size_t)&tile[0];

  v4u g0l = { 1u, lds, (unsigned int)ga, (unsigned int)(ga >> 32) | (2u << 30) };
  v4u g0s = { 1u, lds, (unsigned int)gb, (unsigned int)(gb >> 32) | (2u << 30) };
  // dim0 = 2048 floats/row, dim1 = 8 rows; tile == tensor; stride0 = 2048
  v8i g1 = { (int)(2u << 16),                       // data_size = 4B
             (int)((2048u & 0xFFFFu) << 16),        // tensor_dim0[15:0]
             (int)((2048u >> 16) | (8u << 16)),     // tensor_dim0[31:16] | tensor_dim1
             (int)(2048u << 16),                    // tensor_dim1 hi | tile_dim0
             (int)8,                                // tile_dim1 | tile_dim2=0
             (int)2048,                             // tensor_dim0_stride lo
             0, 0 };
  v4i z4 = { 0, 0, 0, 0 };
  v8i z8 = { 0, 0, 0, 0, 0, 0, 0, 0 };
  __builtin_amdgcn_tensor_load_to_lds(g0l, g1, z4, z4, z8, 0);
  __builtin_amdgcn_s_wait_tensorcnt(0);
  __builtin_amdgcn_tensor_store_from_lds(g0s, g1, z4, z4, z8, 0);
  __builtin_amdgcn_s_wait_tensorcnt(0);
}

// ---------------- conversion / init kernels ----------------
__global__ void cvt_kernel(const float* __restrict__ in, uint16_t* __restrict__ out, int n) {
  for (int i = blockIdx.x * blockDim.x + threadIdx.x; i < n; i += gridDim.x * blockDim.x)
    out[i] = f2bf(in[i]);
}
// in (K x Nin) -> out (Npad x K) transposed+padded bf16
__global__ void cvt_t_kernel(const float* __restrict__ in, uint16_t* __restrict__ out,
                             int K, int Nin, int Npad) {
  int total = Npad * K;
  for (int i = blockIdx.x * blockDim.x + threadIdx.x; i < total; i += gridDim.x * blockDim.x) {
    int n = i / K, k = i - n * K;
    out[i] = (n < Nin) ? f2bf(in[(size_t)k * Nin + n]) : (uint16_t)0;
  }
}
__global__ void addv_kernel(const float* a, const float* b, float* o, int n) {
  int i = blockIdx.x * blockDim.x + threadIdx.x;
  if (i < n) o[i] = a[i] + b[i];
}
__global__ void fill_f32_kernel(float* p, float v, int n) {
  int i = blockIdx.x * blockDim.x + threadIdx.x;
  if (i < n) p[i] = v;
}
__global__ void fill_u16_kernel(uint16_t* p, uint16_t v, int n) {
  int i = blockIdx.x * blockDim.x + threadIdx.x;
  if (i < n) p[i] = v;
}

// ---------------- per-step kernels ----------------
__global__ void bn_concat_kernel(const float* __restrict__ x, const float* __restrict__ lrv,
                                 uint16_t* __restrict__ lin, int t) {
  int f = threadIdx.x;                     // 384 threads
  if (f < kXD) {
    float mu = 0.f;
    for (int b = 0; b < kBS; ++b) mu += x[((size_t)b * kT + t) * kXD + f];
    mu *= (1.0f / kBS);
    float var = 0.f;
    for (int b = 0; b < kBS; ++b) {
      float d = x[((size_t)b * kT + t) * kXD + f] - mu; var += d * d;
    }
    float inv = rsqrtf(var * (1.0f / kBS) + 1e-5f);
    for (int b = 0; b < kBS; ++b)
      lin[b * kIN0 + f] = f2bf((x[((size_t)b * kT + t) * kXD + f] - mu) * inv);
  } else {
    int i = f - kXD;                       // read-vector columns
    for (int b = 0; b < kBS; ++b) lin[b * kIN0 + kXD + i] = f2bf(lrv[b * (kW * kR) + i]);
  }
}

__global__ void lstm_cell_kernel(const float* __restrict__ gates, float* __restrict__ c,
                                 uint16_t* __restrict__ flatbf, int layer) {
  int idx = blockIdx.x * blockDim.x + threadIdx.x;
  if (idx >= kBS * 2 * kHD) return;
  int b = idx >> 10, j = idx & 1023, dir = j >> 9, jj = j & 511;
  const float* g = gates + (size_t)b * (2 * kG4) + dir * kG4;
  float ig = sigm(g[jj]);
  float fg = sigm(g[kHD + jj]);
  float gg = tanhf(g[2 * kHD + jj]);
  float og = sigm(g[3 * kHD + jj]);
  float* cp = c + ((size_t)dir * kBS + b) * kHD + jj;
  float c2 = fg * (*cp) + ig * gg;
  *cp = c2;
  flatbf[(size_t)b * kFLAT + layer * 1024 + dir * kHD + jj] = f2bf(og * tanhf(c2));
}

__global__ void ln_kernel(float* __restrict__ iv) {
  int b = blockIdx.x, tid = threadIdx.x;
  __shared__ float s[256];
  float* row = iv + (size_t)b * kETP;
  float sum = 0.f;
  for (int i = tid; i < kET; i += 256) sum += row[i];
  s[tid] = sum; __syncthreads();
  for (int st = 128; st > 0; st >>= 1) { if (tid < st) s[tid] += s[tid + st]; __syncthreads(); }
  float mu = s[0] / kET; __syncthreads();
  float sq = 0.f;
  for (int i = tid; i < kET; i += 256) { float d = row[i] - mu; sq += d * d; }
  s[tid] = sq; __syncthreads();
  for (int st = 128; st > 0; st >>= 1) { if (tid < st) s[tid] += s[tid + st]; __syncthreads(); }
  float inv = rsqrtf(s[0] / kET + 1e-5f); __syncthreads();
  for (int i = tid; i < kET; i += 256) row[i] = (row[i] - mu) * inv;
}

__global__ void alloc_kernel(const float* __restrict__ iv, const float* __restrict__ lrw,
                             float* __restrict__ u, const float* __restrict__ lww,
                             float* __restrict__ allocw) {
  int b = blockIdx.x, tid = threadIdx.x;                // 512 threads, N=1024
  __shared__ float s_key[kN];
  __shared__ float s_p[kN];
  __shared__ int   s_idx[kN];
  float fg[kR];
#pragma unroll
  for (int r = 0; r < kR; ++r) fg[r] = sigm(iv[(size_t)b * kETP + kOffFg + r]);
  for (int n = tid; n < kN; n += 512) {
    float ret = 1.f;
#pragma unroll
    for (int r = 0; r < kR; ++r) ret *= 1.f - fg[r] * lrw[((size_t)b * kN + n) * kR + r];
    float uo = u[(size_t)b * kN + n], lw = lww[(size_t)b * kN + n];
    float un = (uo + lw - uo * lw) * ret;
    u[(size_t)b * kN + n] = un;
    s_key[n] = un; s_idx[n] = n;
  }
  __syncthreads();
  for (int k = 2; k <= kN; k <<= 1) {
    for (int j = k >> 1; j > 0; j >>= 1) {
      for (int n = tid; n < kN; n += 512) {
        int ixj = n ^ j;
        if (ixj > n) {
          bool up = ((n & k) == 0);
          float a = s_key[n], c = s_key[ixj];
          if ((a > c) == up) {
            s_key[n] = c; s_key[ixj] = a;
            int t2 = s_idx[n]; s_idx[n] = s_idx[ixj]; s_idx[ixj] = t2;
          }
        }
      }
      __syncthreads();
    }
  }
  for (int n = tid; n < kN; n += 512) s_p[n] = s_key[n];
  __syncthreads();
  for (int off = 1; off < kN; off <<= 1) {               // inclusive product scan
    int n0 = tid, n1 = tid + 512;
    float v0 = (n0 >= off) ? s_p[n0 - off] * s_p[n0] : s_p[n0];
    float v1 = (n1 >= off) ? s_p[n1 - off] * s_p[n1] : s_p[n1];
    __syncthreads();
    s_p[n0] = v0; s_p[n1] = v1;
    __syncthreads();
  }
  for (int n = tid; n < kN; n += 512) {
    float cpex = (n == 0) ? 1.f : s_p[n - 1];
    allocw[(size_t)b * kN + s_idx[n]] = (1.f - s_key[n]) * cpex;  // scatter back
  }
}

__global__ void cw_kernel(const float* __restrict__ iv, const float* __restrict__ M,
                          const float* __restrict__ allocw, float* __restrict__ lww) {
  int b = blockIdx.x, tid = threadIdx.x;
  __shared__ float s_k[kW];
  __shared__ float s_red[1024];
  if (tid < kW) async_ld_f32(&s_k[tid], &iv[(size_t)b * kETP + kOffWkey + tid]);
  wait_async0();
  __syncthreads();
  float nk = 0.f;
  for (int w = 0; w < kW; ++w) nk += s_k[w] * s_k[w];
  nk = sqrtf(nk) + kEPS;
  float wbeta = 1.f + softplusf(-iv[(size_t)b * kETP + kOffWbeta]);
  const float* Mrow = M + ((size_t)b * kN + tid) * kW;
  float dot = 0.f, nm = 0.f;
  for (int w = 0; w < kW; ++w) { float m = Mrow[w]; dot += m * s_k[w]; nm += m * m; }
  float score = wbeta * dot / ((sqrtf(nm) + kEPS) * nk);
  s_red[tid] = score; __syncthreads();
  for (int st = 512; st > 0; st >>= 1) { if (tid < st) s_red[tid] = fmaxf(s_red[tid], s_red[tid + st]); __syncthreads(); }
  float mx = s_red[0]; __syncthreads();
  float e = __expf(score - mx);
  s_red[tid] = e; __syncthreads();
  for (int st = 512; st > 0; st >>= 1) { if (tid < st) s_red[tid] += s_red[tid + st]; __syncthreads(); }
  float cw = e / s_red[0];
  float ag = sigm(iv[(size_t)b * kETP + kOffAg]);
  float wg = sigm(iv[(size_t)b * kETP + kOffWg]);
  lww[(size_t)b * kN + tid] = wg * (ag * allocw[(size_t)b * kN + tid] + (1.f - ag) * cw);
}

__global__ void erase_write_kernel(const float* __restrict__ iv, const float* __restrict__ lww,
                                   float* __restrict__ M) {
  int idx = blockIdx.x * blockDim.x + threadIdx.x;
  if (idx >= kBS * kN * kW) return;
  int b = idx >> 16, n = (idx >> 6) & (kN - 1), w = idx & (kW - 1);
  float e  = sigm(iv[(size_t)b * kETP + kOffErase + w]);
  float wv = iv[(size_t)b * kETP + kOffWvec + w];
  float ww = lww[(size_t)b * kN + n];
  M[idx] = M[idx] * (1.f - ww * e) + ww * wv;
}

__global__ void rw_kernel(const float* __restrict__ iv, const float* __restrict__ M,
                          float* __restrict__ lrw) {
  int b = blockIdx.x >> 2, r = blockIdx.x & 3, tid = threadIdx.x;
  __shared__ float s_k[kW];
  __shared__ float s_red[1024];
  if (tid < kW) async_ld_f32(&s_k[tid], &iv[(size_t)b * kETP + kOffRkeys + tid * kR + r]);
  wait_async0();
  __syncthreads();
  float nk = 0.f;
  for (int w = 0; w < kW; ++w) nk += s_k[w] * s_k[w];
  nk = sqrtf(nk) + kEPS;
  float rbeta = 1.f + softplusf(-iv[(size_t)b * kETP + kOffRbeta + r]);
  const float* Mrow = M + ((size_t)b * kN + tid) * kW;
  float dot = 0.f, nm = 0.f;
  for (int w = 0; w < kW; ++w) { float m = Mrow[w]; dot += m * s_k[w]; nm += m * m; }
  float score = rbeta * dot / ((sqrtf(nm) + kEPS) * nk);
  s_red[tid] = score; __syncthreads();
  for (int st = 512; st > 0; st >>= 1) { if (tid < st) s_red[tid] = fmaxf(s_red[tid], s_red[tid + st]); __syncthreads(); }
  float mx = s_red[0]; __syncthreads();
  float e = __expf(score - mx);
  s_red[tid] = e; __syncthreads();
  for (int st = 512; st > 0; st >>= 1) { if (tid < st) s_red[tid] += s_red[tid + st]; __syncthreads(); }
  lrw[((size_t)b * kN + tid) * kR + r] = e / s_red[0];
}

__global__ void rv_kernel(const float* __restrict__ M, const float* __restrict__ lrw,
                          float* __restrict__ lrv, uint16_t* __restrict__ rvbf) {
  int b = blockIdx.x, tid = threadIdx.x;
  int w = tid & 63, seg = tid >> 6;                       // 4 segments of 256 rows
  __shared__ float s[4 * kW * kR];
  float acc[kR] = {0.f, 0.f, 0.f, 0.f};
  for (int n = seg * 256; n < (seg + 1) * 256; ++n) {
    float m = M[((size_t)b * kN + n) * kW + w];
#pragma unroll
    for (int r = 0; r < kR; ++r) acc[r] += m * lrw[((size_t)b * kN + n) * kR + r];
  }
#pragma unroll
  for (int r = 0; r < kR; ++r) s[(seg * kW + w) * kR + r] = acc[r];
  __syncthreads();
  if (seg == 0) {
#pragma unroll
    for (int r = 0; r < kR; ++r) {
      float v = s[(0 * kW + w) * kR + r] + s[(1 * kW + w) * kR + r] +
                s[(2 * kW + w) * kR + r] + s[(3 * kW + w) * kR + r];
      lrv[(size_t)b * kW * kR + w * kR + r] = v;
      rvbf[(size_t)b * kW * kR + w * kR + r] = f2bf(v);
    }
  }
}

__global__ void max_kernel(const float* __restrict__ vt, float* __restrict__ out) {
  int i = blockIdx.x * blockDim.x + threadIdx.x;
  if (i < kBS * kVT) out[i] = fmaxf(out[i], vt[i]);
}

// ---------------- workspace layout ----------------
constexpr size_t A256(size_t x) { return (x + 255) & ~(size_t)255; }
constexpr size_t OFF_WYB   = 0;
constexpr size_t OFF_WEB   = A256(OFF_WYB   + (size_t)kVT  * kFLAT * 2);
constexpr size_t OFF_WRB   = A256(OFF_WEB   + (size_t)kETP * kFLAT * 2);
constexpr size_t OFF_WIH0  = A256(OFF_WRB   + (size_t)kVT  * 256   * 2);
constexpr size_t OFF_WHH0  = A256(OFF_WIH0  + (size_t)2 * kG4 * kIN0 * 2);
constexpr size_t OFF_WIH1  = A256(OFF_WHH0  + (size_t)2 * kG4 * kHD  * 2);
constexpr size_t OFF_WHH1  = A256(OFF_WIH1  + (size_t)2 * kG4 * 1024 * 2);
constexpr size_t OFF_B0    = A256(OFF_WHH1  + (size_t)2 * kG4 * kHD  * 2);
constexpr size_t OFF_B1    = A256(OFF_B0    + (size_t)2 * kG4 * 4);
constexpr size_t OFF_M     = A256(OFF_B1    + (size_t)2 * kG4 * 4);
constexpr size_t OFF_U     = A256(OFF_M     + (size_t)kBS * kN * kW * 4);
constexpr size_t OFF_LRW   = A256(OFF_U     + (size_t)kBS * kN * 4);
constexpr size_t OFF_LWW   = A256(OFF_LRW   + (size_t)kBS * kN * kR * 4);
constexpr size_t OFF_LRV   = A256(OFF_LWW   + (size_t)kBS * kN * 4);
constexpr size_t OFF_CST   = A256(OFF_LRV   + (size_t)kBS * kW * kR * 4);
constexpr size_t OFF_FLAT  = A256(OFF_CST   + (size_t)4 * kBS * kHD * 4);
constexpr size_t OFF_LIN   = A256(OFF_FLAT  + (size_t)kBS * kFLAT * 2);
constexpr size_t OFF_GATES = A256(OFF_LIN   + (size_t)kBS * kIN0 * 2);
constexpr size_t OFF_VTB   = A256(OFF_GATES + (size_t)kBS * 2 * kG4 * 4);
constexpr size_t OFF_IVB   = A256(OFF_VTB   + (size_t)kBS * kVT * 4);
constexpr size_t OFF_ALLOC = A256(OFF_IVB   + (size_t)kBS * kETP * 4);
constexpr size_t OFF_RVBF  = A256(OFF_ALLOC + (size_t)kBS * kN * 4);

static void launch_gemm(hipStream_t s, const uint16_t* A, int lda, const uint16_t* B,
                        const float* bias, float* C, int ldc, int K, int Nout,
                        int nValid, int accum) {
  int waves = Nout >> 4;
  int thr = 256;
  int blocks = (waves * 32 + thr - 1) / thr;
  wmma_gemm_kernel<<<blocks, thr, 0, s>>>(A, lda, B, bias, C, ldc, K, Nout, nValid, accum);
}

extern "C" void kernel_launch(void* const* d_in, const int* in_sizes, int n_in,
                              void* d_out, int out_size, void* d_ws, size_t ws_size,
                              hipStream_t stream) {
  const float* x       = (const float*)d_in[0];
  const float* memory0 = (const float*)d_in[1];
  const float* Wy      = (const float*)d_in[2];
  const float* WE      = (const float*)d_in[3];
  const float* Wr      = (const float*)d_in[4];
  const float* Wih0    = (const float*)d_in[5];
  const float* Whh0    = (const float*)d_in[6];
  const float* bih0    = (const float*)d_in[7];
  const float* bhh0    = (const float*)d_in[8];
  const float* Wih1    = (const float*)d_in[9];
  const float* Whh1    = (const float*)d_in[10];
  const float* bih1    = (const float*)d_in[11];
  const float* bhh1    = (const float*)d_in[12];

  char* ws = (char*)d_ws;
  uint16_t* WYB  = (uint16_t*)(ws + OFF_WYB);
  uint16_t* WEB  = (uint16_t*)(ws + OFF_WEB);
  uint16_t* WRB  = (uint16_t*)(ws + OFF_WRB);
  uint16_t* WIH0 = (uint16_t*)(ws + OFF_WIH0);
  uint16_t* WHH0 = (uint16_t*)(ws + OFF_WHH0);
  uint16_t* WIH1 = (uint16_t*)(ws + OFF_WIH1);
  uint16_t* WHH1 = (uint16_t*)(ws + OFF_WHH1);
  float*    B0   = (float*)(ws + OFF_B0);
  float*    B1   = (float*)(ws + OFF_B1);
  float*    M    = (float*)(ws + OFF_M);
  float*    U    = (float*)(ws + OFF_U);
  float*    LRW  = (float*)(ws + OFF_LRW);
  float*    LWW  = (float*)(ws + OFF_LWW);
  float*    LRV  = (float*)(ws + OFF_LRV);
  float*    CST  = (float*)(ws + OFF_CST);
  uint16_t* FLAT = (uint16_t*)(ws + OFF_FLAT);
  uint16_t* LIN  = (uint16_t*)(ws + OFF_LIN);
  float*    GAT  = (float*)(ws + OFF_GATES);
  float*    VTB  = (float*)(ws + OFF_VTB);
  float*    IVB  = (float*)(ws + OFF_IVB);
  float*    ALC  = (float*)(ws + OFF_ALLOC);
  uint16_t* RVBF = (uint16_t*)(ws + OFF_RVBF);
  float*    OUT  = (float*)d_out;

  // ---- weight conversion (bf16; Wy/WE/Wr transposed to (N,K) so k-pairs pack) ----
  cvt_kernel<<<1024, 256, 0, stream>>>(Wih0, WIH0, 2 * kG4 * kIN0);
  cvt_kernel<<<1024, 256, 0, stream>>>(Whh0, WHH0, 2 * kG4 * kHD);
  cvt_kernel<<<1024, 256, 0, stream>>>(Wih1, WIH1, 2 * kG4 * 1024);
  cvt_kernel<<<1024, 256, 0, stream>>>(Whh1, WHH1, 2 * kG4 * kHD);
  cvt_t_kernel<<<2048, 256, 0, stream>>>(Wy, WYB, kFLAT, kVT, kVT);
  cvt_t_kernel<<<512,  256, 0, stream>>>(WE, WEB, kFLAT, kET, kETP);
  cvt_t_kernel<<<512,  256, 0, stream>>>(Wr, WRB, 256, kVT, kVT);
  addv_kernel<<<16, 256, 0, stream>>>(bih0, bhh0, B0, 2 * kG4);
  addv_kernel<<<16, 256, 0, stream>>>(bih1, bhh1, B1, 2 * kG4);

  // ---- state init (M copied via Tensor Data Mover) ----
  tdm_copy_kernel<<<128, 32, 0, stream>>>(memory0, M);
  fill_f32_kernel<<<128, 256, 0, stream>>>(U, 0.f, kBS * kN);
  fill_f32_kernel<<<512, 256, 0, stream>>>(LRW, 0.f, kBS * kN * kR);
  fill_f32_kernel<<<128, 256, 0, stream>>>(LWW, 0.f, kBS * kN);
  fill_f32_kernel<<<32, 256, 0, stream>>>(LRV, 0.f, kBS * kW * kR);
  fill_f32_kernel<<<256, 256, 0, stream>>>(CST, 0.f, 4 * kBS * kHD);
  fill_u16_kernel<<<256, 256, 0, stream>>>(FLAT, 0, kBS * kFLAT);
  fill_u16_kernel<<<32, 256, 0, stream>>>(RVBF, 0, kBS * kW * kR);
  fill_f32_kernel<<<256, 256, 0, stream>>>(OUT, -3.4e38f, kBS * kVT);

  for (int t = 0; t < kT; ++t) {
    bn_concat_kernel<<<1, kIN0, 0, stream>>>(x, LRV, LIN, t);

    // ---- LSTM layer 0 (flat cols [0,512)=hf0, [512,1024)=hb0 hold h_{t-1}) ----
    launch_gemm(stream, LIN, kIN0, WIH0, B0, GAT, 2 * kG4, kIN0, 2 * kG4, 2 * kG4, 0);
    launch_gemm(stream, FLAT + 0,    kFLAT, WHH0,               nullptr, GAT,        2 * kG4, kHD, kG4, kG4, 1);
    launch_gemm(stream, FLAT + kHD,  kFLAT, WHH0 + kG4 * kHD,   nullptr, GAT + kG4,  2 * kG4, kHD, kG4, kG4, 1);
    lstm_cell_kernel<<<(kBS * 2 * kHD + 255) / 256, 256, 0, stream>>>(GAT, CST, FLAT, 0);

    // ---- LSTM layer 1 (input = flat cols [0,1024); h_{t-1} in cols [1024,2048)) ----
    launch_gemm(stream, FLAT, kFLAT, WIH1, B1, GAT, 2 * kG4, 1024, 2 * kG4, 2 * kG4, 0);
    launch_gemm(stream, FLAT + 1024,       kFLAT, WHH1,             nullptr, GAT,       2 * kG4, kHD, kG4, kG4, 1);
    launch_gemm(stream, FLAT + 1024 + kHD, kFLAT, WHH1 + kG4 * kHD, nullptr, GAT + kG4, 2 * kG4, kHD, kG4, kG4, 1);
    lstm_cell_kernel<<<(kBS * 2 * kHD + 255) / 256, 256, 0, stream>>>(GAT, CST + 2 * kBS * kHD, FLAT, 1);

    // ---- head projections ----
    launch_gemm(stream, FLAT, kFLAT, WYB, nullptr, VTB, kVT, kFLAT, kVT, kVT, 0);
    launch_gemm(stream, FLAT, kFLAT, WEB, nullptr, IVB, kETP, kFLAT, kETP, kET, 0);
    ln_kernel<<<kBS, 256, 0, stream>>>(IVB);

    // ---- memory module ----
    alloc_kernel<<<kBS, 512, 0, stream>>>(IVB, LRW, U, LWW, ALC);
    cw_kernel<<<kBS, 1024, 0, stream>>>(IVB, M, ALC, LWW);
    erase_write_kernel<<<(kBS * kN * kW + 255) / 256, 256, 0, stream>>>(IVB, LWW, M);
    rw_kernel<<<kBS * kR, 1024, 0, stream>>>(IVB, M, LRW);
    rv_kernel<<<kBS, 256, 0, stream>>>(M, LRW, LRV, RVBF);

    // ---- output: yt = vt + rv@Wr ; running max ----
    launch_gemm(stream, RVBF, kW * kR, WRB, nullptr, VTB, kVT, kW * kR, kVT, kVT, 1);
    max_kernel<<<(kBS * kVT + 255) / 256, 256, 0, stream>>>(VTB, OUT);
  }
}
